// GroupQueryAttention_61546881351758
// MI455X (gfx1250) — matrix operand order repaired and checked by reading, not compiled, not run on hardware
//
#include <hip/hip_runtime.h>
#include <hip/hip_bf16.h>

// GQA forward for MI455X (gfx1250), bf16 WMMA (v_wmma_f32_16x16x32_bf16),
// f32 accumulation, flash-attention style online softmax.
//
// B=2, T=2048, C=1024, H=16, G=4, D=64, GS=4, BT=4096.

typedef __bf16 bf16;
typedef __attribute__((ext_vector_type(16))) __bf16 v16bf;
typedef __attribute__((ext_vector_type(8)))  __bf16 v8bf;
typedef __attribute__((ext_vector_type(4)))  __bf16 v4bf;
typedef __attribute__((ext_vector_type(8)))  float  v8f;

__device__ __forceinline__ v8f zero8() {
  v8f z = {0.f, 0.f, 0.f, 0.f, 0.f, 0.f, 0.f, 0.f};
  return z;
}

// A-fragment (16x32 bf16, MxK): lane m = lane%16, elements K = {8h+0..7, 8h+16..23}
// p = row base pointer (already includes (m0+lane%16)*stride + k0), h = lane>>4
__device__ __forceinline__ v16bf frag_a(const bf16* p, int h) {
  union { v16bf v; v8bf q[2]; } u;
  u.q[0] = *(const v8bf*)(p + 8 * h);
  u.q[1] = *(const v8bf*)(p + 8 * h + 16);
  return u.v;
}

// B-fragment (32x16 bf16, KxN): lane n = lane%16, elements K = {8h+0..7, 8h+16..23}
// p = "column source" row base (row-major over K), contiguous in K.
__device__ __forceinline__ v16bf frag_b(const bf16* p, int h) {
  union { v16bf v; v8bf q[2]; } u;
  u.q[0] = *(const v8bf*)(p + 8 * h);
  u.q[1] = *(const v8bf*)(p + 8 * h + 16);
  return u.v;
}

__device__ __forceinline__ v8f wmma_bf16(v16bf a, v16bf b, v8f c) {
  return __builtin_amdgcn_wmma_f32_16x16x32_bf16(
      /*neg_a=*/false, a, /*neg_b=*/false, b,
      /*c_mod=*/(short)0, c, /*reuse_a=*/false, /*reuse_b=*/false);
}

// ---------------------------------------------------------------------------
// fp32 -> bf16 conversion (vectorized x4)
// ---------------------------------------------------------------------------
__global__ void __launch_bounds__(256) f32_to_bf16_kernel(
    const float* __restrict__ s, bf16* __restrict__ d, int n) {
  int i = (blockIdx.x * 256 + threadIdx.x) * 4;
  if (i < n) {
    float4 f = *(const float4*)(s + i);
    v4bf o;
    o[0] = (bf16)f.x; o[1] = (bf16)f.y; o[2] = (bf16)f.z; o[3] = (bf16)f.w;
    *(v4bf*)(d + i) = o;
  }
}

// ---------------------------------------------------------------------------
// Y[M,N] = X[M,K] * W[N,K]^T  (bf16 in, bf16 out)
// Computed transposed per tile: C^T = W_tile * X_tile^T so that C-layout
// register rows (n) map to the contiguous output dimension -> v8bf stores.
// Each wave: 64x64 output tile = 4 n-subtiles x 4 m-subtiles (16 accumulators,
// 16 WMMAs per 16 b128 loads per k-step -> 16 B of L2 traffic per WMMA).
// ---------------------------------------------------------------------------
__global__ void __launch_bounds__(256) gemm_nt_bf16(
    const bf16* __restrict__ X, const bf16* __restrict__ W,
    bf16* __restrict__ Y, int M, int N, int K) {
  const int lane = threadIdx.x & 31;
  const int wave = threadIdx.x >> 5;
  const int h = lane >> 4;
  const int ln = lane & 15;
  const int tilesN = N >> 6;
  const int tilesM = M >> 6;
  const int wid = blockIdx.x * 8 + wave;
  if (wid >= tilesN * tilesM) return;
  const int n0 = (wid % tilesN) << 6;
  const int m0 = (wid / tilesN) << 6;

  v8f acc[4][4];
#pragma unroll
  for (int i = 0; i < 4; ++i)
#pragma unroll
    for (int j = 0; j < 4; ++j) acc[i][j] = zero8();

  const bf16* wrow = W + (size_t)(n0 + ln) * K;
  const bf16* xrow = X + (size_t)(m0 + ln) * K;
  const size_t r16 = (size_t)16 * K;

  for (int k0 = 0; k0 < K; k0 += 32) {
    __builtin_prefetch(wrow + k0 + 256, 0, 3);  // near-scope global_prefetch_b8
    __builtin_prefetch(xrow + k0 + 256, 0, 3);
    v16bf a[4];
#pragma unroll
    for (int j = 0; j < 4; ++j) a[j] = frag_a(wrow + j * r16 + k0, h);
#pragma unroll
    for (int i = 0; i < 4; ++i) {
      v16bf b = frag_b(xrow + i * r16 + k0, h);
#pragma unroll
      for (int j = 0; j < 4; ++j) acc[i][j] = wmma_bf16(a[j], b, acc[i][j]);
    }
  }

  // acc[i][j][r] = Y^T[n0+16j+8h+r][m0+16i+ln] -> row m, 8 contiguous cols
#pragma unroll
  for (int i = 0; i < 4; ++i) {
    bf16* yrow = Y + (size_t)(m0 + 16 * i + ln) * N + n0 + 8 * h;
#pragma unroll
    for (int j = 0; j < 4; ++j) {
      v8bf o;
#pragma unroll
      for (int r = 0; r < 8; ++r) o[r] = (bf16)acc[i][j][r];
      *(v8bf*)(yrow + 16 * j) = o;
    }
  }
}

// ---------------------------------------------------------------------------
// Final projection: Y[M,N](f32) = X[M,K](bf16) * W[N,K]^T + bias[N]
// ---------------------------------------------------------------------------
__global__ void __launch_bounds__(256) gemm_nt_f32(
    const bf16* __restrict__ X, const bf16* __restrict__ W,
    const float* __restrict__ bias, float* __restrict__ Y,
    int M, int N, int K) {
  const int lane = threadIdx.x & 31;
  const int wave = threadIdx.x >> 5;
  const int h = lane >> 4;
  const int ln = lane & 15;
  const int tilesN = N >> 6;
  const int tilesM = M >> 6;
  const int wid = blockIdx.x * 8 + wave;
  if (wid >= tilesN * tilesM) return;
  const int n0 = (wid % tilesN) << 6;
  const int m0 = (wid / tilesN) << 6;

  v8f acc[4][4];
#pragma unroll
  for (int i = 0; i < 4; ++i)
#pragma unroll
    for (int j = 0; j < 4; ++j) acc[i][j] = zero8();

  const bf16* wrow = W + (size_t)(n0 + ln) * K;
  const bf16* xrow = X + (size_t)(m0 + ln) * K;
  const size_t r16 = (size_t)16 * K;

  for (int k0 = 0; k0 < K; k0 += 32) {
    __builtin_prefetch(wrow + k0 + 256, 0, 3);
    __builtin_prefetch(xrow + k0 + 256, 0, 3);
    v16bf a[4];
#pragma unroll
    for (int j = 0; j < 4; ++j) a[j] = frag_a(wrow + j * r16 + k0, h);
#pragma unroll
    for (int i = 0; i < 4; ++i) {
      v16bf b = frag_b(xrow + i * r16 + k0, h);
#pragma unroll
      for (int j = 0; j < 4; ++j) acc[i][j] = wmma_bf16(a[j], b, acc[i][j]);
    }
  }

#pragma unroll
  for (int i = 0; i < 4; ++i) {
    float* yrow = Y + (size_t)(m0 + 16 * i + ln) * N + n0 + 8 * h;
#pragma unroll
    for (int j = 0; j < 4; ++j) {
      v8f bv = *(const v8f*)(bias + n0 + 16 * j + 8 * h);
      v8f o = acc[i][j] + bv;
      *(v8f*)(yrow + 16 * j) = o;
    }
  }
}

// ---------------------------------------------------------------------------
// Flash GQA attention.
// Q: [B*T, 1024] (col = h*64+d), K/V: [B*T, 256] (col = g*64+d),
// O: [B*T, 1024] bf16.
// Grid: (B*H, T/64). Block: 128 threads = 4 waves; wave w owns q rows
// [tq_base + 16w, +16). Computes S^T = K*Q^T (softmax stats per-lane over
// t = lane%16), P^T round-trips LDS, O^T += V^T * P^T with V staged
// transposed in LDS.
// ---------------------------------------------------------------------------
__global__ void __launch_bounds__(128) flash_gqa(
    const bf16* __restrict__ Q, const bf16* __restrict__ Km,
    const bf16* __restrict__ Vm, bf16* __restrict__ O) {
  __shared__ bf16 Vt[64][40];      // V^T tile: [d][s], padded stride
  __shared__ bf16 Pt[4][16][40];   // per-wave P tile: [t][s]

  const int lane = threadIdx.x & 31;
  const int wave = threadIdx.x >> 5;
  const int h1 = lane >> 4;
  const int ln = lane & 15;

  const int bh = blockIdx.x;       // 0..31
  const int b  = bh >> 4;          // / H
  const int hh = bh & 15;          // % H
  const int g  = hh >> 2;          // head -> group (GS=4)
  const int tq_base = blockIdx.y * 64;
  const int tq0 = tq_base + wave * 16;
  const int tg  = tq0 + ln;        // this lane's query index t

  // Preload Q as two B-fragments (d = 0..31, 32..63)
  const bf16* qrow = Q + (size_t)(b * 2048 + tq0 + ln) * 1024 + hh * 64;
  const v16bf qb0 = frag_b(qrow, h1);
  const v16bf qb1 = frag_b(qrow + 32, h1);

  const bf16* kbase = Km + (size_t)(b * 2048) * 256 + g * 64;
  const bf16* vbase = Vm + (size_t)(b * 2048) * 256 + g * 64;

  float m = -1e30f, l = 0.f;
  v8f o0 = zero8(), o1 = zero8(), o2 = zero8(), o3 = zero8();

  const int nblocks = (tq_base + 64) >> 5;  // causal extent for this block
  for (int bi = 0; bi < nblocks; ++bi) {
    const int s0 = bi << 5;

    __syncthreads();  // previous Vt reads complete
    {
      // cooperative staged transpose of V[s0..s0+31][0..63] -> Vt[d][s]
      const int sl = threadIdx.x >> 2;        // 0..31
      const int dq = (threadIdx.x & 3) << 4;  // 0,16,32,48
      const bf16* vr = vbase + (size_t)(s0 + sl) * 256 + dq;
      v8bf x0 = *(const v8bf*)vr;
      v8bf x1 = *(const v8bf*)(vr + 8);
#pragma unroll
      for (int i = 0; i < 8; ++i) {
        Vt[dq + i][sl]     = x0[i];
        Vt[dq + 8 + i][sl] = x1[i];
      }
    }
    __syncthreads();

    if (s0 < tq0 + 16) {  // wave-uniform: EXEC stays all-ones for WMMA
      // S^T = K * Q^T : two 16-row s tiles, K-dim = D = 64 (2 steps)
      const bf16* krow0 = kbase + (size_t)(s0 + ln) * 256;
      const bf16* krow1 = kbase + (size_t)(s0 + 16 + ln) * 256;
      v8f st0 = zero8(), st1 = zero8();
      st0 = wmma_bf16(frag_a(krow0, h1),      qb0, st0);
      st0 = wmma_bf16(frag_a(krow0 + 32, h1), qb1, st0);
      st1 = wmma_bf16(frag_a(krow1, h1),      qb0, st1);
      st1 = wmma_bf16(frag_a(krow1 + 32, h1), qb1, st1);

      // scale + causal mask; per-lane row stats (t = ln fixed per lane)
      const float sc = 0.125f;  // 1/sqrt(64)
      float z0[8], z1[8], zmax = -1e30f;
#pragma unroll
      for (int r = 0; r < 8; ++r) {
        int sg = s0 + 8 * h1 + r;
        float z = (sg <= tg) ? st0[r] * sc : -1e30f;
        z0[r] = z; zmax = fmaxf(zmax, z);
        sg = s0 + 16 + 8 * h1 + r;
        z = (sg <= tg) ? st1[r] * sc : -1e30f;
        z1[r] = z; zmax = fmaxf(zmax, z);
      }
      zmax = fmaxf(zmax, __shfl_xor(zmax, 16));
      const float mnew  = fmaxf(m, zmax);
      const float alpha = __expf(m - mnew);
      m = mnew;

      float ps = 0.f;
      bf16 p0[8], p1[8];
#pragma unroll
      for (int r = 0; r < 8; ++r) {
        float p = __expf(z0[r] - m); ps += p; p0[r] = (bf16)p;
        p       = __expf(z1[r] - m); ps += p; p1[r] = (bf16)p;
      }
      ps += __shfl_xor(ps, 16);
      l = l * alpha + ps;
      o0 *= alpha; o1 *= alpha; o2 *= alpha; o3 *= alpha;

      // P^T (C-layout: lane holds col t=ln, rows s=8h1+r / 16+8h1+r) -> LDS [t][s]
#pragma unroll
      for (int r = 0; r < 8; ++r) {
        Pt[wave][ln][8 * h1 + r]      = p0[r];
        Pt[wave][ln][16 + 8 * h1 + r] = p1[r];
      }
      // Reload P as B-fragment (lane col t=ln, K=s={8h1+0..7, 16+8h1+0..7})
      const v16bf pb = frag_b(&Pt[wave][ln][0], h1);

      // O^T += V^T * P^T : 4 d-tiles, A-fragments from transposed LDS V
      o0 = wmma_bf16(frag_a(&Vt[ln][0], h1),      pb, o0);
      o1 = wmma_bf16(frag_a(&Vt[16 + ln][0], h1), pb, o1);
      o2 = wmma_bf16(frag_a(&Vt[32 + ln][0], h1), pb, o2);
      o3 = wmma_bf16(frag_a(&Vt[48 + ln][0], h1), pb, o3);
    }
  }

  // epilogue: O^T[d][t], lane col t=ln, rows d = 16*mt + 8h1 + r (contiguous)
  const float inv = 1.0f / l;
  bf16* orow = O + (size_t)(b * 2048 + tq0 + ln) * 1024 + hh * 64;
  auto store_o = [&](v8f o, int mt) {
    v8bf pk;
#pragma unroll
    for (int r = 0; r < 8; ++r) pk[r] = (bf16)(o[r] * inv);
    *(v8bf*)(orow + 16 * mt + 8 * h1) = pk;
  };
  store_o(o0, 0); store_o(o1, 1); store_o(o2, 2); store_o(o3, 3);
}

// ---------------------------------------------------------------------------
// Launch
// ---------------------------------------------------------------------------
extern "C" void kernel_launch(void* const* d_in, const int* in_sizes, int n_in,
                              void* d_out, int out_size, void* d_ws, size_t ws_size,
                              hipStream_t stream) {
  const float* x  = (const float*)d_in[0];  // [2,2048,1024]
  const float* Wq = (const float*)d_in[1];  // [16,64,1024] -> rows (h*64+d)
  const float* Wk = (const float*)d_in[2];  // [4,64,1024]  -> rows (g*64+d)
  const float* Wv = (const float*)d_in[3];
  const float* Wp = (const float*)d_in[4];  // [1024,1024]
  const float* bp = (const float*)d_in[5];  // [1024]
  float* out = (float*)d_out;               // [4096,1024] f32

  // workspace layout (bf16 elements), ~33 MB total
  bf16* ws  = (bf16*)d_ws;
  bf16* xb  = ws; ws += 4096 * 1024;
  bf16* Wqb = ws; ws += 1024 * 1024;
  bf16* Wkb = ws; ws +=  256 * 1024;
  bf16* Wvb = ws; ws +=  256 * 1024;
  bf16* Wpb = ws; ws += 1024 * 1024;
  bf16* qb  = ws; ws += 4096 * 1024;
  bf16* kb  = ws; ws += 4096 * 256;
  bf16* vb  = ws; ws += 4096 * 256;
  bf16* ob  = ws; ws += 4096 * 1024;

  auto cvt = [&](const float* s, bf16* d, int n) {
    f32_to_bf16_kernel<<<dim3((n / 4 + 255) / 256), dim3(256), 0, stream>>>(s, d, n);
  };
  cvt(x,  xb,  4096 * 1024);
  cvt(Wq, Wqb, 1024 * 1024);
  cvt(Wk, Wkb,  256 * 1024);
  cvt(Wv, Wvb,  256 * 1024);
  cvt(Wp, Wpb, 1024 * 1024);

  // Projections: waves = (N/64)*(M/64), 8 waves per 256-thread block
  gemm_nt_bf16<<<dim3(128), dim3(256), 0, stream>>>(xb, Wqb, qb, 4096, 1024, 1024);
  gemm_nt_bf16<<<dim3(32),  dim3(256), 0, stream>>>(xb, Wkb, kb, 4096, 256, 1024);
  gemm_nt_bf16<<<dim3(32),  dim3(256), 0, stream>>>(xb, Wvb, vb, 4096, 256, 1024);

  // Attention: (B*H, T/64) blocks of 128 threads
  flash_gqa<<<dim3(32, 32), dim3(128), 0, stream>>>(qb, kb, vb, ob);

  // Output projection + bias (f32 out)
  gemm_nt_f32<<<dim3(128), dim3(256), 0, stream>>>(ob, Wpb, bp, out, 4096, 1024, 1024);
}
